// PINN_Model_steady_64295660421650
// MI455X (gfx1250) — compile-verified, matrix operand order checked by reading, not run on hardware
//
#include <hip/hip_runtime.h>

// ---------------------------------------------------------------------------
// PINN forward + Jacobian + Hessian via forward-over-forward AD, batched as
// bf16 WMMA GEMMs on gfx1250 (MI455X).
//
// Per 16-point tile: state = 15 components x 16 points = 240 rows of width 128.
// Each hidden layer: Z[240x128] = S[240x128] @ W[128x128]  (v_wmma 16x16x32 bf16,
// f32 accumulate), then pointwise tanh chain rule recombines components.
// Wave w owns N-tile w (B fragments hoisted to registers), sweeps all 15 M-tiles.
// ---------------------------------------------------------------------------

typedef __attribute__((ext_vector_type(16))) __bf16 v16bf;
typedef __attribute__((ext_vector_type(8)))  __bf16 v8bf;
typedef __attribute__((ext_vector_type(8)))  float  v8f;

#define D        128
#define NLAYERS  5
#define NC       15          // 1 value + 4 Jacobian + 10 Hessian components
#define PTS_WG   16
#define NROWS    (NC * PTS_WG)   // 240
#define SSTRIDE  136         // bf16 elems per row: 272 B, 16B aligned
#define ZSTRIDE  132         // f32 elems per row

// LDS carve (bytes, all 16B aligned)
#define OFF_S    0
#define SZ_S     (NROWS * SSTRIDE * 2)          // 65280
#define OFF_Z    (OFF_S + SZ_S)                 // 65280
#define SZ_Z     (NROWS * ZSTRIDE * 4)          // 126720
#define OFF_WT   (OFF_Z + SZ_Z)                 // 192000
#define SZ_WT    (D * SSTRIDE * 2)              // 34816
#define OFF_WO   (OFF_WT + SZ_WT)               // 226816
#define SZ_WO    (D * 4 * 4)                    // 2048
#define OFF_BO   (OFF_WO + SZ_WO)               // 228864
#define SZ_BO    16
#define OFF_R    (OFF_BO + SZ_BO)               // 228880
#define SZ_R     (NC * 4 * PTS_WG * 4)          // 3840
#define SMEM_TOTAL (OFF_R + SZ_R)               // 232720 B < 320 KB WGP budget

__global__ __launch_bounds__(256)
void pinn_fwd_jac_hess_kernel(const float* __restrict__ x,
                              const float* __restrict__ y,
                              const float* __restrict__ z,
                              const float* __restrict__ t,
                              const float* __restrict__ W0,
                              const float* __restrict__ b0,
                              const float* __restrict__ Wh,
                              const float* __restrict__ bh,
                              const float* __restrict__ Wout,
                              const float* __restrict__ bout,
                              const float* __restrict__ actc,
                              float* __restrict__ out,
                              int npts)
{
    extern __shared__ char smem[];
    __bf16* S  = (__bf16*)(smem + OFF_S);   // bf16 A-operand state
    float*  Z  = (float*) (smem + OFF_Z);   // f32 pre-act / master state
    __bf16* WT = (__bf16*)(smem + OFF_WT);  // transposed weights: WT[n][k]=W[k][n]
    float*  Wo = (float*) (smem + OFF_WO);
    float*  Bo = (float*) (smem + OFF_BO);
    float*  Rb = (float*) (smem + OFF_R);

    const int tid  = threadIdx.x;
    const int base = blockIdx.x * PTS_WG;
    const float a  = actc[0];

    // --- preload output-layer weights ---
    for (int i = tid; i < D * 4; i += 256) Wo[i] = Wout[i];
    if (tid < 4) Bo[tid] = bout[tid];

    // --- layer 0: 4 -> 128 (scalar; K=4) + tanh chain rule, init S ---
    for (int e = tid; e < PTS_WG * D; e += 256) {
        const int p = e >> 7, d = e & 127;
        const int gp = base + p;
        const float X0 = x[gp], X1 = y[gp], X2 = z[gp], X3 = 2.0f * t[gp] - 1.0f;
        const float w0d = W0[0 * D + d], w1d = W0[1 * D + d];
        const float w2d = W0[2 * D + d], w3d = W0[3 * D + d];
        const float zh = b0[d] + X0 * w0d + X1 * w1d + X2 * w2d + X3 * w3d;
        float zJ[4] = { w0d, w1d, w2d, 2.0f * w3d };   // input-normalization scales
        const float th = tanhf(a * zh);
        const float s2 = 1.0f - th * th;
        const float g1 = a * s2;
        const float g2 = -2.0f * a * a * th * s2;
        S[(0 * PTS_WG + p) * SSTRIDE + d] = (__bf16)th;
        #pragma unroll
        for (int i = 0; i < 4; ++i)
            S[((1 + i) * PTS_WG + p) * SSTRIDE + d] = (__bf16)(g1 * zJ[i]);
        int q = 0;
        #pragma unroll
        for (int i = 0; i < 4; ++i)
            #pragma unroll
            for (int j = i; j < 4; ++j, ++q)
                S[((5 + q) * PTS_WG + p) * SSTRIDE + d] = (__bf16)(g2 * zJ[i] * zJ[j]);
    }
    __syncthreads();

    const int wave  = tid >> 5;         // 8 waves == 8 N-tiles of 16 columns
    const int lane  = tid & 31;
    const int lrow  = lane & 15;        // row (A) / col (B) / col (D) within tile
    const int lhalf = lane >> 4;        // k-half select per 16-bit operand layout
    const int n0    = wave * 16;        // this wave's fixed output-column tile

    for (int layer = 0; layer < NLAYERS; ++layer) {
        const float* Wl = Wh + (size_t)layer * D * D;

        // transpose-convert weights into LDS: WT[n][k] = (bf16) W[k][n]
        for (int i = tid; i < D * D; i += 256) {
            const int k = i >> 7, n = i & 127;
            WT[n * SSTRIDE + k] = (__bf16)Wl[i];
        }
        // prefetch next layer's weights into cache (global_prefetch_b8)
        if (layer + 1 < NLAYERS) {
            const char* nxt = (const char*)(Wh + (size_t)(layer + 1) * D * D);
            __builtin_prefetch(nxt + (size_t)tid * 256, 0, 0);
        }
        __syncthreads();

        // --- hoist this wave's B fragments (invariant across all 15 M-tiles) ---
        v16bf bv[4];
        #pragma unroll
        for (int kb = 0; kb < 4; ++kb) {
            const int k0 = kb * 32 + lhalf * 8;
            v8bf blo = *(const v8bf*)&WT[(n0 + lrow) * SSTRIDE + k0];
            v8bf bhi = *(const v8bf*)&WT[(n0 + lrow) * SSTRIDE + k0 + 16];
            bv[kb] = __builtin_shufflevector(blo, bhi,
                0,1,2,3,4,5,6,7,8,9,10,11,12,13,14,15);
        }

        // --- GEMM: Z[240x128] = S @ W ; 15 M-tiles per wave, K=128 (4 steps) ---
        for (int mt = 0; mt < NC; ++mt) {
            const int m0 = mt * 16;
            v8f c = {};
            #pragma unroll
            for (int kb = 0; kb < 4; ++kb) {
                const int k0 = kb * 32 + lhalf * 8;
                v8bf alo = *(const v8bf*)&S[(m0 + lrow) * SSTRIDE + k0];
                v8bf ahi = *(const v8bf*)&S[(m0 + lrow) * SSTRIDE + k0 + 16];
                v16bf av = __builtin_shufflevector(alo, ahi,
                    0,1,2,3,4,5,6,7,8,9,10,11,12,13,14,15);
                c = __builtin_amdgcn_wmma_f32_16x16x32_bf16(
                        false, av, false, bv[kb], (short)0, c, false, false);
            }
            #pragma unroll
            for (int r = 0; r < 8; ++r)
                Z[(m0 + r + lhalf * 8) * ZSTRIDE + n0 + lrow] = c[r];
        }
        __syncthreads();

        // pointwise tanh chain rule; bias folded in here (value component only)
        for (int e = tid; e < PTS_WG * D; e += 256) {
            const int p = e >> 7, d = e & 127;
            const float zh = Z[(0 * PTS_WG + p) * ZSTRIDE + d] + bh[layer * D + d];
            float zJ[4], zH[10];
            #pragma unroll
            for (int i = 0; i < 4; ++i)
                zJ[i] = Z[((1 + i) * PTS_WG + p) * ZSTRIDE + d];
            #pragma unroll
            for (int q = 0; q < 10; ++q)
                zH[q] = Z[((5 + q) * PTS_WG + p) * ZSTRIDE + d];
            const float th = tanhf(a * zh);
            const float s2 = 1.0f - th * th;
            const float g1 = a * s2;
            const float g2 = -2.0f * a * a * th * s2;
            S[(0 * PTS_WG + p) * SSTRIDE + d] = (__bf16)th;
            Z[(0 * PTS_WG + p) * ZSTRIDE + d] = th;
            #pragma unroll
            for (int i = 0; i < 4; ++i) {
                const float nv = g1 * zJ[i];
                S[((1 + i) * PTS_WG + p) * SSTRIDE + d] = (__bf16)nv;
                Z[((1 + i) * PTS_WG + p) * ZSTRIDE + d] = nv;
            }
            int q = 0;
            #pragma unroll
            for (int i = 0; i < 4; ++i)
                #pragma unroll
                for (int j = i; j < 4; ++j, ++q) {
                    const float nv = g1 * zH[q] + g2 * zJ[i] * zJ[j];
                    S[((5 + q) * PTS_WG + p) * SSTRIDE + d] = (__bf16)nv;
                    Z[((5 + q) * PTS_WG + p) * ZSTRIDE + d] = nv;
                }
        }
        __syncthreads();
    }

    // --- output layer 128 -> 4 (scalar; N=4): Rb[comp][o][p] ---
    for (int e = tid; e < NC * 4 * PTS_WG; e += 256) {
        const int comp = e >> 6;
        const int o = (e >> 4) & 3;
        const int p = e & 15;
        float acc = (comp == 0) ? Bo[o] : 0.0f;
        const float* srow = &Z[(comp * PTS_WG + p) * ZSTRIDE];
        for (int d = 0; d < D; ++d) acc += srow[d] * Wo[d * 4 + o];
        Rb[e] = acc;
    }
    __syncthreads();

    // --- Navier-Stokes residual combine, one lane per point ---
    if (tid < PTS_WG) {
        const int p = tid, gp = base + p;
        #define RB(c, o) Rb[(c) * 64 + (o) * 16 + p]
        const float u = RB(0, 0), v = RB(0, 1), w = RB(0, 2);
        const float ux = RB(1, 0), uy = RB(2, 0), uz = RB(3, 0), ut = RB(4, 0);
        const float vx = RB(1, 1), vy = RB(2, 1), vz = RB(3, 1), vt = RB(4, 1);
        const float wx = RB(1, 2), wy = RB(2, 2), wz = RB(3, 2), wt = RB(4, 2);
        const float px = RB(1, 3), py = RB(2, 3), pz = RB(3, 3);
        // pair indices: (0,0)=5 (0,1)=6 (0,2)=7 (0,3)=8 (1,1)=9 (1,2)=10
        //               (1,3)=11 (2,2)=12 (2,3)=13 (3,3)=14
        const float tau_x = 2.0f * RB(5, 0) + RB(9, 0) + RB(6, 1) + RB(12, 0) + RB(7, 2);
        const float tau_y = RB(6, 0) + RB(5, 1) + 2.0f * RB(9, 1) + RB(12, 1) + RB(10, 2);
        const float tau_z = RB(7, 0) + RB(5, 2) + RB(10, 1) + RB(9, 2) + 2.0f * RB(12, 2);
        out[0 * npts + gp] = ux + vy + wz;
        out[1 * npts + gp] = ut + (u * ux + v * uy + w * uz) + px - tau_x;
        out[2 * npts + gp] = vt + (u * vx + v * vy + w * vz) + py - tau_y;
        out[3 * npts + gp] = wt + (u * wx + v * wy + w * wz) + pz - tau_z + 98.9f;
        #undef RB
    }
}

extern "C" void kernel_launch(void* const* d_in, const int* in_sizes, int n_in,
                              void* d_out, int out_size, void* d_ws, size_t ws_size,
                              hipStream_t stream) {
    (void)n_in; (void)out_size; (void)d_ws; (void)ws_size;
    const float* x    = (const float*)d_in[0];
    const float* y    = (const float*)d_in[1];
    const float* z    = (const float*)d_in[2];
    const float* t    = (const float*)d_in[3];
    const float* W0   = (const float*)d_in[4];
    const float* b0   = (const float*)d_in[5];
    const float* Wh   = (const float*)d_in[6];
    const float* bh   = (const float*)d_in[7];
    const float* Wout = (const float*)d_in[8];
    const float* bout = (const float*)d_in[9];
    const float* ac   = (const float*)d_in[10];
    float* out = (float*)d_out;

    const int npts   = in_sizes[0];
    const int blocks = npts / PTS_WG;

    hipFuncSetAttribute((const void*)pinn_fwd_jac_hess_kernel,
                        hipFuncAttributeMaxDynamicSharedMemorySize,
                        (int)SMEM_TOTAL);
    pinn_fwd_jac_hess_kernel<<<blocks, 256, SMEM_TOTAL, stream>>>(
        x, y, z, t, W0, b0, Wh, bh, Wout, bout, ac, out, npts);
}
